// QuantLinearInt4_7292854468914
// MI455X (gfx1250) — matrix operand order repaired and checked by reading, not compiled, not run on hardware
//
#include <hip/hip_runtime.h>

typedef _Float16 h2   __attribute__((ext_vector_type(2)));
typedef _Float16 v16h __attribute__((ext_vector_type(16)));
typedef float    v8f  __attribute__((ext_vector_type(8)));
typedef unsigned int       u32;
typedef unsigned long long u64;
typedef int v4i __attribute__((__vector_size__(4 * sizeof(int))));

constexpr int M = 32, K = 8192, N = 8192;
constexpr int KSPLIT    = 4;
constexpr int KSLICE    = K / KSPLIT;     // 2048 K per block
constexpr int KC        = 64;             // K per main-loop iteration
constexpr int ITERS     = KSLICE / KC;    // 32
constexpr int ROWSTRIDE = 72;             // padded halves per A row in LDS (bank-conflict free)

#if defined(__HIP_DEVICE_COMPILE__) && __has_builtin(__builtin_amdgcn_global_load_async_to_lds_b128)
#define USE_ASYNC_LDS 1
#else
#define USE_ASYNC_LDS 0
#endif

__device__ __forceinline__ void wait_asynccnt0() {
#if __has_builtin(__builtin_amdgcn_s_wait_asynccnt)
  __builtin_amdgcn_s_wait_asynccnt(0);
#else
  asm volatile("s_wait_asynccnt 0x0" ::: "memory");
#endif
}

// Dequant two int4 nibbles -> packed fp16 pair.
// perm places byte j of w in [7:0] and byte j of (w>>4) in [23:16];
// mask low nibbles and xor 0x6408 (fused by clang into one v_bitop3):
// flips sign bit (q+8) + sets fp16 exponent (1024+); subtract 1032 -> exact q.
__device__ __forceinline__ h2 dq_pair(u32 w, u32 w4, u32 sel) {
  u32 p = (__builtin_amdgcn_perm(w4, w, sel) & 0x000F000Fu) ^ 0x64086408u;
  h2 v = __builtin_bit_cast(h2, p);
  h2 c = {(_Float16)1032.0f, (_Float16)1032.0f};
  return v - c;
}

__global__ __launch_bounds__(128)
void qlin_main(const _Float16* __restrict__ x,
               const int*      __restrict__ bpk,
               float*          __restrict__ part) {
  __shared__ alignas(16) _Float16 sA[2][32 * ROWSTRIDE];

  const int t    = threadIdx.x;
  const int lane = t & 31;
  const int wave = t >> 5;
  const int hi   = (lane >> 4) & 1;   // 0: K0-15 half, 1: K16-31 half
  const int lcol = lane & 15;

  const int n     = (blockIdx.x * 4 + wave) * 16 + lcol;  // output column this lane owns
  const int kbase = blockIdx.y * KSLICE;

  // B stream: one u64 = 16 nibbles along K for row n
  const u64* bp = (const u64*)bpk + (size_t)n * (K / 16) + (kbase / 16) + hi;

  const _Float16* xs = x + kbase;

  v8f acc0 = {};  // rows 0..15
  v8f acc1 = {};  // rows 16..31

  // Cooperative A staging: 128 threads x 2 chunks x 16B = 32x64 halves.
  auto stage = [&](int it, int sel) {
#pragma unroll
    for (int i = 0; i < 2; ++i) {
      int chunk = t + 128 * i;
      int row = chunk >> 3;
      int c8  = (chunk & 7) << 3;
      const _Float16* src = xs + (size_t)row * K + it * KC + c8;
      _Float16*       dst = &sA[sel][row * ROWSTRIDE + c8];
#if USE_ASYNC_LDS
      // CDNA5 async DMA: global -> LDS without VGPR roundtrip (ASYNCcnt).
      __builtin_amdgcn_global_load_async_to_lds_b128(
          (__attribute__((address_space(1))) v4i*)src,
          (__attribute__((address_space(3))) v4i*)dst, 0, 0);
#else
      *(uint4*)dst = *(const uint4*)src;
#endif
    }
  };

  stage(0, 0);

  for (int it = 0; it < ITERS; ++it) {
    // B loads for this iteration (the HBM-bound stream) — issue early
    u64 raw0 = bp[it * 4];
    u64 raw1 = bp[it * 4 + 2];
    __builtin_prefetch((const void*)(bp + it * 4 + 32), 0, 1);  // ~8 iters ahead

#if USE_ASYNC_LDS
    wait_asynccnt0();   // this wave's staging of buffer (it&1) has landed in LDS
#endif
    __syncthreads();    // all waves' staging done; other buffer free for overwrite

    if (it + 1 < ITERS) stage(it + 1, (it + 1) & 1);

    const _Float16* A = sA[it & 1];

#pragma unroll
    for (int ks = 0; ks < 2; ++ks) {
      u64 raw = ks ? raw1 : raw0;
      u32 w0 = (u32)raw, w1 = (u32)(raw >> 32);
      u32 w04 = w0 >> 4, w14 = w1 >> 4;

      union { h2 p[8]; v16h v; } b;
      b.p[0] = dq_pair(w0, w04, 0x0C040C00u);
      b.p[1] = dq_pair(w0, w04, 0x0C050C01u);
      b.p[2] = dq_pair(w0, w04, 0x0C060C02u);
      b.p[3] = dq_pair(w0, w04, 0x0C070C03u);
      b.p[4] = dq_pair(w1, w14, 0x0C040C00u);
      b.p[5] = dq_pair(w1, w14, 0x0C050C01u);
      b.p[6] = dq_pair(w1, w14, 0x0C060C02u);
      b.p[7] = dq_pair(w1, w14, 0x0C070C03u);

      // A fragments in contiguous K order matching B: lanes 0-15 v0-7=K0-15;
      // lanes 16-31 v0-7=K16-31 (offset hi*16).
      const int kofs = ks * 32 + hi * 16;
      union { uint4 q[2]; v16h v; } a0, a1;
      a0.q[0] = *(const uint4*)(A + lcol * ROWSTRIDE + kofs);
      a0.q[1] = *(const uint4*)(A + lcol * ROWSTRIDE + kofs + 8);
      a1.q[0] = *(const uint4*)(A + (lcol + 16) * ROWSTRIDE + kofs);
      a1.q[1] = *(const uint4*)(A + (lcol + 16) * ROWSTRIDE + kofs + 8);

      acc0 = __builtin_amdgcn_wmma_f32_16x16x32_f16(false, a0.v, false, b.v,
                                                    (short)0, acc0, false, false);
      acc1 = __builtin_amdgcn_wmma_f32_16x16x32_f16(false, a1.v, false, b.v,
                                                    (short)0, acc1, false, false);
    }
  }

  // Write f32 partials: part[kslice][m][n]. C layout: VGPR r = row r (+8 for hi lanes).
  union { v8f v; float f[8]; } r0, r1;
  r0.v = acc0; r1.v = acc1;
  float* pb = part + (size_t)blockIdx.y * M * N;
#pragma unroll
  for (int r = 0; r < 8; ++r) {
    int m0 = r + hi * 8;
    pb[(size_t)m0 * N + n]        = r0.f[r];
    pb[(size_t)(m0 + 16) * N + n] = r1.f[r];
  }
}

__global__ __launch_bounds__(256)
void qlin_reduce(const float*    __restrict__ part,
                 const _Float16* __restrict__ scales,
                 const _Float16* __restrict__ bias,
                 _Float16*       __restrict__ out) {
  int idx = blockIdx.x * 256 + threadIdx.x;   // over M*N
  int nn  = idx & (N - 1);
  float s = part[idx] + part[(size_t)M * N + idx] +
            part[2 * (size_t)M * N + idx] + part[3 * (size_t)M * N + idx];
  out[idx] = (_Float16)(s * (float)scales[nn] + (float)bias[nn]);
}

extern "C" void kernel_launch(void* const* d_in, const int* in_sizes, int n_in,
                              void* d_out, int out_size, void* d_ws, size_t ws_size,
                              hipStream_t stream) {
  (void)in_sizes; (void)n_in; (void)out_size; (void)ws_size;
  const _Float16* x      = (const _Float16*)d_in[0];
  const int*      bpk    = (const int*)d_in[1];
  const _Float16* scales = (const _Float16*)d_in[2];
  const _Float16* bias   = (const _Float16*)d_in[3];
  _Float16*       out    = (_Float16*)d_out;
  float*          part   = (float*)d_ws;   // KSPLIT * M * N * 4B = 4 MB

  dim3 grid(N / 64, KSPLIT);   // 128 x 4 blocks
  qlin_main<<<grid, 128, 0, stream>>>(x, bpk, part);
  qlin_reduce<<<(M * N) / 256, 256, 0, stream>>>(part, scales, bias, out);
}
